// DotProductAttention_21466246545654
// MI455X (gfx1250) — matrix operand order repaired and checked by reading, not compiled, not run on hardware
//
#include <hip/hip_runtime.h>

typedef __attribute__((ext_vector_type(16))) _Float16 v16h;
typedef __attribute__((ext_vector_type(8)))  _Float16 v8h;
typedef __attribute__((ext_vector_type(4)))  _Float16 v4h;
typedef __attribute__((ext_vector_type(8)))  float    v8f;
typedef __attribute__((ext_vector_type(4)))  float    v4f;
typedef int v4i_vs __attribute__((vector_size(16)));   // matches builtin param type

#define B_    16
#define S_    2048
#define D_    128
#define QTILE 128
#define KTILE 64
#define NWAVE 8
#define PAD   8

// ---- CDNA5 async global->LDS path (guarded; falls back to sync staging) ----
#if defined(__gfx1250__) && \
    __has_builtin(__builtin_amdgcn_global_load_async_to_lds_b128) && \
    __has_builtin(__builtin_amdgcn_s_wait_asynccnt)
#define USE_ASYNC 1
#else
#define USE_ASYNC 0
#endif

#define AS1 __attribute__((address_space(1)))
#define AS3 __attribute__((address_space(3)))

#if USE_ASYNC
__device__ __forceinline__ void async_copy16(const float* g, float* lds) {
    // 16 bytes per lane, wave-wide: GLOBAL_LOAD_ASYNC_TO_LDS_B128 (ASYNCcnt)
    __builtin_amdgcn_global_load_async_to_lds_b128(
        (AS1 v4i_vs*)(__UINTPTR_TYPE__)g,  // global flat == AS1 numerically; drops const
        (AS3 v4i_vs*)lds,                  // addrspacecast strips LDS aperture
        0, 0);
}
#endif

__launch_bounds__(256, 1)
__global__ void fa_fwd_f16wmma(const float* __restrict__ Q,
                               const float* __restrict__ K,
                               const float* __restrict__ V,
                               float* __restrict__ O)
{
    // K tile row-major: score B-fragment reads are two 8-f16 runs 16 apart.
    __shared__ __attribute__((aligned(32))) _Float16 sK [KTILE][D_ + PAD];     // 64 x 136
    // V tile transposed [depth][key]: second-GEMM B-fragments contiguous.
    __shared__ __attribute__((aligned(32))) _Float16 sVt[D_][KTILE + PAD];     // 128 x 72
    // Per-wave P staging: C-layout -> A-layout relayout for the second GEMM.
    __shared__ __attribute__((aligned(32))) _Float16 sP [NWAVE][16][KTILE + PAD];
#if USE_ASYNC
    // Raw f32 staging for the async pipeline (tile is contiguous in memory).
    __shared__ __attribute__((aligned(16))) float stgK[KTILE * D_];            // 32 KB
    __shared__ __attribute__((aligned(16))) float stgV[KTILE * D_];            // 32 KB
#endif

    const int tid  = threadIdx.x;
    const int wave = tid >> 5;
    const int lane = tid & 31;
    const int half = lane >> 4;   // which 16-lane half of the wave
    const int ncol = lane & 15;   // N index (C/B layouts) / M index (A layout)

    const int b  = blockIdx.x / (S_ / QTILE);
    const int q0 = (blockIdx.x % (S_ / QTILE)) * QTILE;

    const float qscale = 0.08838834764831845f; // 1/sqrt(128)

    // ---------------- Q -> f16 A fragments (pre-scaled) ----------------
    // A 16-bit layout: lanes 0-15 M=lane, K = {8*half+e, 16+8*half+e}
    v16h aQ[4];
    {
        const float* qrow = Q + ((size_t)b * S_ + (size_t)(q0 + wave * 16 + ncol)) * D_;
        #pragma unroll
        for (int kk = 0; kk < 4; ++kk) {
            #pragma unroll
            for (int e = 0; e < 8; ++e)
                aQ[kk][e]     = (_Float16)(qrow[kk * 32 +      8 * half + e] * qscale);
            #pragma unroll
            for (int e = 0; e < 8; ++e)
                aQ[kk][8 + e] = (_Float16)(qrow[kk * 32 + 16 + 8 * half + e] * qscale);
        }
    }

    // ---------------- accumulators / softmax state ----------------
    v8f accO[8];
    #pragma unroll
    for (int i = 0; i < 8; ++i) accO[i] = v8f{};
    float m_run[8], l_run[8];
    #pragma unroll
    for (int r = 0; r < 8; ++r) { m_run[r] = -3.0e38f; l_run[r] = 0.0f; }

    const float* kbat = K + (size_t)b * S_ * D_;
    const float* vbat = V + (size_t)b * S_ * D_;

#if USE_ASYNC
    // Prologue: kick off tile 0 (8 x b128 per thread per tensor = whole 32KB tile)
    #pragma unroll
    for (int it = 0; it < 8; ++it) {
        const int i = (it * 256 + tid) * 4;
        async_copy16(kbat + i, stgK + i);
        async_copy16(vbat + i, stgV + i);
    }
#endif

    for (int kt = 0; kt < S_ / KTILE; ++kt) {
        const float* kbase = kbat + (size_t)kt * KTILE * D_;
        const float* vbase = vbat + (size_t)kt * KTILE * D_;

#if USE_ASYNC
        // Own async loads landed -> barrier makes every wave's data visible.
        __builtin_amdgcn_s_wait_asynccnt(0);
        __syncthreads();
        // Convert staged f32 -> f16 tiles (K row-major, V transposed).
        #pragma unroll
        for (int it = 0; it < 8; ++it) {
            const int i   = (it * 256 + tid) * 4;
            const int key = i / D_;
            const int d   = i % D_;
            v4f kf = *(const v4f*)(stgK + i);
            v4h kh;
            kh[0] = (_Float16)kf[0]; kh[1] = (_Float16)kf[1];
            kh[2] = (_Float16)kf[2]; kh[3] = (_Float16)kf[3];
            *(v4h*)&sK[key][d] = kh;
            v4f vf = *(const v4f*)(stgV + i);
            sVt[d + 0][key] = (_Float16)vf[0];
            sVt[d + 1][key] = (_Float16)vf[1];
            sVt[d + 2][key] = (_Float16)vf[2];
            sVt[d + 3][key] = (_Float16)vf[3];
        }
        __syncthreads();  // staging buffers free; f16 tiles ready
        // Overlap: fetch tile kt+1 while computing tile kt.
        if (kt + 1 < S_ / KTILE) {
            const float* knext = kbase + (size_t)KTILE * D_;
            const float* vnext = vbase + (size_t)KTILE * D_;
            #pragma unroll
            for (int it = 0; it < 8; ++it) {
                const int i = (it * 256 + tid) * 4;
                async_copy16(knext + i, stgK + i);
                async_copy16(vnext + i, stgV + i);
            }
        }
#else
        // Synchronous staging: global -> VGPR -> cvt -> LDS
        #pragma unroll
        for (int it = 0; it < (KTILE * D_) / (4 * 256); ++it) { // 8 iters
            const int i   = (it * 256 + tid) * 4;
            const int key = i / D_;
            const int d   = i % D_;
            v4f kf = *(const v4f*)(kbase + (size_t)key * D_ + d);
            v4h kh;
            kh[0] = (_Float16)kf[0]; kh[1] = (_Float16)kf[1];
            kh[2] = (_Float16)kf[2]; kh[3] = (_Float16)kf[3];
            *(v4h*)&sK[key][d] = kh;
            v4f vf = *(const v4f*)(vbase + (size_t)key * D_ + d);
            sVt[d + 0][key] = (_Float16)vf[0];
            sVt[d + 1][key] = (_Float16)vf[1];
            sVt[d + 2][key] = (_Float16)vf[2];
            sVt[d + 3][key] = (_Float16)vf[3];
        }
        if (kt + 1 < S_ / KTILE) {
            __builtin_prefetch(kbase + (size_t)KTILE * D_ + (size_t)tid * 32, 0, 3);
            __builtin_prefetch(vbase + (size_t)KTILE * D_ + (size_t)tid * 32, 0, 3);
        }
        __syncthreads();
#endif

        // ---- S = Q * K^T  (4 key blocks x 4 depth chunks of WMMA) ----
        v8f sblk[4];
        #pragma unroll
        for (int nb = 0; nb < 4; ++nb) {
            v8f acc = v8f{};
            const _Float16* krow = &sK[nb * 16 + ncol][8 * half];
            #pragma unroll
            for (int kk = 0; kk < 4; ++kk) {
                const _Float16* kp = krow + kk * 32;
                v8h lo = *(const v8h*)kp;
                v8h hi = *(const v8h*)(kp + 16);
                v16h bK = __builtin_shufflevector(lo, hi,
                    0,1,2,3,4,5,6,7,8,9,10,11,12,13,14,15);
                acc = __builtin_amdgcn_wmma_f32_16x16x32_f16(
                          false, aQ[kk], false, bK, (short)0, acc, false, false);
            }
            sblk[nb] = acc;
        }

        // ---- online softmax (row r lives at m = r + 8*half) ----
        #pragma unroll
        for (int r = 0; r < 8; ++r) {
            float vmax = fmaxf(fmaxf(sblk[0][r], sblk[1][r]),
                               fmaxf(sblk[2][r], sblk[3][r]));
            vmax = fmaxf(vmax, __shfl_xor(vmax, 1, 32));
            vmax = fmaxf(vmax, __shfl_xor(vmax, 2, 32));
            vmax = fmaxf(vmax, __shfl_xor(vmax, 4, 32));
            vmax = fmaxf(vmax, __shfl_xor(vmax, 8, 32));
            const float mnew = fmaxf(m_run[r], vmax);
            const float corr = __expf(m_run[r] - mnew);
            m_run[r] = mnew;
            float rsum = 0.0f;
            #pragma unroll
            for (int nb = 0; nb < 4; ++nb) {
                float p = __expf(sblk[nb][r] - mnew);
                sblk[nb][r] = p;
                rsum += p;
            }
            rsum += __shfl_xor(rsum, 1, 32);
            rsum += __shfl_xor(rsum, 2, 32);
            rsum += __shfl_xor(rsum, 4, 32);
            rsum += __shfl_xor(rsum, 8, 32);
            l_run[r] = l_run[r] * corr + rsum;
            #pragma unroll
            for (int db = 0; db < 8; ++db) accO[db][r] *= corr;
        }

        // ---- stage P (f16) for A-layout reads ----
        #pragma unroll
        for (int nb = 0; nb < 4; ++nb)
            #pragma unroll
            for (int r = 0; r < 8; ++r)
                sP[wave][r + 8 * half][nb * 16 + ncol] = (_Float16)sblk[nb][r];

        // ---- O += P * V  (2 key chunks x 8 depth blocks of WMMA) ----
        #pragma unroll
        for (int c = 0; c < 2; ++c) {
            const _Float16* prow = &sP[wave][ncol][c * 32 + 8 * half];
            v8h plo = *(const v8h*)prow;
            v8h phi = *(const v8h*)(prow + 16);
            v16h aP = __builtin_shufflevector(plo, phi,
                0,1,2,3,4,5,6,7,8,9,10,11,12,13,14,15);
            #pragma unroll
            for (int db = 0; db < 8; ++db) {
                const _Float16* vp = &sVt[db * 16 + ncol][c * 32 + 8 * half];
                v8h vlo = *(const v8h*)vp;
                v8h vhi = *(const v8h*)(vp + 16);
                v16h bV = __builtin_shufflevector(vlo, vhi,
                    0,1,2,3,4,5,6,7,8,9,10,11,12,13,14,15);
                accO[db] = __builtin_amdgcn_wmma_f32_16x16x32_f16(
                               false, aP, false, bV, (short)0, accO[db], false, false);
            }
        }
#if !USE_ASYNC
        __syncthreads();
#endif
    }

    // ---------------- epilogue: O /= l ----------------
    #pragma unroll
    for (int r = 0; r < 8; ++r) {
        const float inv = 1.0f / l_run[r];
        float* orow = O + ((size_t)b * S_ + (size_t)(q0 + wave * 16 + r + 8 * half)) * D_;
        #pragma unroll
        for (int db = 0; db < 8; ++db)
            orow[db * 16 + ncol] = accO[db][r] * inv;
    }
}

extern "C" void kernel_launch(void* const* d_in, const int* in_sizes, int n_in,
                              void* d_out, int out_size, void* d_ws, size_t ws_size,
                              hipStream_t stream) {
    (void)in_sizes; (void)n_in; (void)d_ws; (void)ws_size; (void)out_size;
    const float* q = (const float*)d_in[0];
    const float* k = (const float*)d_in[1];
    const float* v = (const float*)d_in[2];
    float* o = (float*)d_out;
    dim3 grid(B_ * (S_ / QTILE));   // 256 blocks
    dim3 block(256);                // 8 wave32
    fa_fwd_f16wmma<<<grid, block, 0, stream>>>(q, k, v, o);
}
